// GAT_15865609191698
// MI455X (gfx1250) — matrix operand order repaired and checked by reading, not compiled, not run on hardware
//
#include <hip/hip_runtime.h>
#include <cstdint>
#include <cstddef>

// ---------------------------------------------------------------------------
// GAT forward for gfx1250: bf16 WMMA GEMMs (v_wmma_f32_16x16x32_bf16) +
// atomic-based segment softmax / scatter-add for the attention aggregation.
// GEMM uses a 32x64 wave tile: 2 A-frags x 4 B-frags -> 8 WMMAs per K=32 step.
// ---------------------------------------------------------------------------

typedef __attribute__((ext_vector_type(16))) __bf16 v16bf;
typedef __attribute__((ext_vector_type(8)))  float  v8f;

union FragU { v16bf v; uint4 q[2]; };
union AccU  { v8f  v; float f[8]; };

__device__ __forceinline__ unsigned short f2bf(float f) {
    unsigned int u = __float_as_uint(f);
    u += 0x7FFFu + ((u >> 16) & 1u);            // round-to-nearest-even
    return (unsigned short)(u >> 16);
}
__device__ __forceinline__ float bfbits2f(unsigned int b) {
    return __uint_as_float(b << 16);
}

// ---------------- weight prep ----------------------------------------------
// out[n,k] = bf16( W[n, k] + W[n, k+Khalf] )   (folds the tile(x,(1,2)) input)
__global__ void k_fold_w(const float* __restrict__ W, unsigned short* __restrict__ out,
                         int Nrows, int Khalf) {
    int i = blockIdx.x * blockDim.x + threadIdx.x;
    if (i >= Nrows * Khalf) return;
    int n = i / Khalf, k = i % Khalf;
    float v = W[(size_t)n * 2 * Khalf + k] + W[(size_t)n * 2 * Khalf + Khalf + k];
    out[(size_t)n * Khalf + k] = f2bf(v);
}

__global__ void k_cvt_bf16(const float* __restrict__ in, unsigned short* __restrict__ out,
                           long long n) {
    long long i = blockIdx.x * (long long)blockDim.x + threadIdx.x;
    if (i < n) out[i] = f2bf(in[i]);
}

__global__ void k_fill0(unsigned int* __restrict__ p, long long n) {
    long long i = blockIdx.x * (long long)blockDim.x + threadIdx.x;
    if (i < n) p[i] = 0u;
}

// ---------------- WMMA GEMM: C[M,N] = A[M,K](bf16) * W[N,K](bf16)^T --------
// flags: 1 = add bias[n], 2 = relu, 4 = write bf16 (else f32)
// Wave tile 32x64: A frags for rows [mBase, mBase+16), [mBase+16, mBase+32);
// 4 B frags shared by both row-halves -> 8 WMMAs per 12x16B loads.
#define GF_BIAS 1
#define GF_RELU 2
#define GF_BF16 4

__global__ void __launch_bounds__(128)
k_gemm_bf16(const unsigned short* __restrict__ A,
            const unsigned short* __restrict__ W,
            const float* __restrict__ bias,
            float* __restrict__ outF,
            unsigned short* __restrict__ outB,
            int M, int N, int K, int flags)
{
    const int lane = threadIdx.x & 31;
    const int wv   = threadIdx.x >> 5;          // 0..3
    const int half = lane >> 4;                 // 0/1
    const int r16  = lane & 15;
    const int nBase = blockIdx.x * 64;
    const int mBase = (blockIdx.y * 4 + wv) * 32;
    if (mBase >= M) return;                     // wave-uniform guard (EXEC stays full)

    // A fragments: lane holds row, K pattern {half*8+0..7, 16+half*8+0..7}
    const unsigned short* aRow0 = A + (size_t)(mBase + r16)      * K + half * 8;
    const unsigned short* aRow1 = A + (size_t)(mBase + 16 + r16) * K + half * 8;
    // B fragments: lane holds W-row (col of B) n, K pattern {half*8+0..7, 16+half*8+0..7}
    const unsigned short* bR[4];
#pragma unroll
    for (int t = 0; t < 4; ++t)
        bR[t] = W + (size_t)(nBase + 16 * t + r16) * K + half * 8;

    AccU acc0[4], acc1[4];
#pragma unroll
    for (int t = 0; t < 4; ++t) {
        acc0[t].v = (v8f){0,0,0,0,0,0,0,0};
        acc1[t].v = (v8f){0,0,0,0,0,0,0,0};
    }

    for (int k = 0; k < K; k += 32) {
        FragU a0, a1;
        a0.q[0] = *(const uint4*)(aRow0 + k);
        a0.q[1] = *(const uint4*)(aRow0 + k + 16);
        a1.q[0] = *(const uint4*)(aRow1 + k);
        a1.q[1] = *(const uint4*)(aRow1 + k + 16);
        if (k + 128 < K) {                       // global_prefetch_b8
            __builtin_prefetch(aRow0 + k + 128, 0, 3);
            __builtin_prefetch(aRow1 + k + 128, 0, 3);
        }
#pragma unroll
        for (int t = 0; t < 4; ++t) {
            FragU b;
            b.q[0] = *(const uint4*)(bR[t] + k);
            b.q[1] = *(const uint4*)(bR[t] + k + 16);
            acc0[t].v = __builtin_amdgcn_wmma_f32_16x16x32_bf16(
                false, a0.v, false, b.v, (short)0, acc0[t].v, false, false);
            acc1[t].v = __builtin_amdgcn_wmma_f32_16x16x32_bf16(
                false, a1.v, false, b.v, (short)0, acc1[t].v, false, false);
        }
    }

    const bool hasBias = flags & GF_BIAS;
    const bool doRelu  = flags & GF_RELU;
    const bool asBf    = flags & GF_BF16;
#pragma unroll
    for (int t = 0; t < 4; ++t) {
        int n = nBase + 16 * t + r16;
        float bn = hasBias ? bias[n] : 0.0f;
#pragma unroll
        for (int j = 0; j < 8; ++j) {           // D: M = mBase(+16) + 8*half + j, N = n
            float v0 = acc0[t].f[j] + bn;
            float v1 = acc1[t].f[j] + bn;
            if (doRelu) { v0 = fmaxf(v0, 0.0f); v1 = fmaxf(v1, 0.0f); }
            size_t i0 = (size_t)(mBase      + 8 * half + j) * N + n;
            size_t i1 = (size_t)(mBase + 16 + 8 * half + j) * N + n;
            if (asBf) { outB[i0] = f2bf(v0); outB[i1] = f2bf(v1); }
            else      { outF[i0] = v0;       outF[i1] = v1; }
        }
    }
}

// ---------------- attention scalars: a_s[n,h] = <h[n,h,:], attS[h,:]> ------
__global__ void k_att_dots(const unsigned short* __restrict__ h,
                           const float* __restrict__ attS,
                           const float* __restrict__ attD,
                           float* __restrict__ aS, float* __restrict__ aD,
                           int Ntot, int H, int C)
{
    int tid = blockIdx.x * blockDim.x + threadIdx.x;
    if (tid >= Ntot * H) return;
    int n = tid / H, hh = tid % H;
    const unsigned short* hp = h + (size_t)n * H * C + (size_t)hh * C;
    const float* sp = attS + hh * C;
    const float* dp = attD + hh * C;
    float accS = 0.f, accD = 0.f;
    for (int c = 0; c < C; c += 8) {
        uint4 q = *(const uint4*)(hp + c);
        unsigned int w[4] = {q.x, q.y, q.z, q.w};
#pragma unroll
        for (int j = 0; j < 4; ++j) {
            float lo = bfbits2f(w[j] & 0xFFFFu);
            float hi = bfbits2f(w[j] >> 16);
            accS = fmaf(lo, sp[c + 2*j],     accS);
            accS = fmaf(hi, sp[c + 2*j + 1], accS);
            accD = fmaf(lo, dp[c + 2*j],     accD);
            accD = fmaf(hi, dp[c + 2*j + 1], accD);
        }
    }
    aS[tid] = accS; aD[tid] = accD;
}

// ---------------- edge pass 1: e = leakyrelu(as[src]+ad[dst]); segment max -
__device__ __forceinline__ unsigned int f2key(float f) {
    unsigned int u = __float_as_uint(f);
    return (u & 0x80000000u) ? ~u : (u | 0x80000000u);   // monotonic in f
}
__device__ __forceinline__ float key2f(unsigned int k) {
    return __uint_as_float((k & 0x80000000u) ? (k ^ 0x80000000u) : ~k);
}

__global__ void k_edge_max(const float* __restrict__ aS, const float* __restrict__ aD,
                           const int* __restrict__ srcA, const int* __restrict__ dstA,
                           int E, int Ntot, int H,
                           float* __restrict__ eBuf, unsigned int* __restrict__ mKey)
{
    int tid = blockIdx.x * blockDim.x + threadIdx.x;
    int tot = (E + Ntot) * H;
    if (tid >= tot) return;
    int i = tid / H, hh = tid % H;
    int s = (i < E) ? srcA[i] : (i - E);
    int d = (i < E) ? dstA[i] : (i - E);
    float e = aS[s * H + hh] + aD[d * H + hh];
    e = (e > 0.f) ? e : 0.2f * e;                // leaky relu, slope 0.2
    eBuf[tid] = e;
    atomicMax(&mKey[d * H + hh], f2key(e));
}

// ---------------- edge pass 2: ex = exp(e - m[dst]); denom += ex -----------
__global__ void k_edge_expsum(const int* __restrict__ srcA, const int* __restrict__ dstA,
                              int E, int Ntot, int H,
                              const unsigned int* __restrict__ mKey,
                              float* __restrict__ eBuf, float* __restrict__ den)
{
    int tid = blockIdx.x * blockDim.x + threadIdx.x;
    int tot = (E + Ntot) * H;
    if (tid >= tot) return;
    int i = tid / H, hh = tid % H;
    int d = (i < E) ? dstA[i] : (i - E);
    float m  = key2f(mKey[d * H + hh]);
    float ex = __expf(eBuf[tid] - m);
    eBuf[tid] = ex;
    atomicAdd(&den[d * H + hh], ex);
}

// ---------------- edge pass 3: agg[dst,h,:] += alpha * h[src,h,:] ----------
// one wave per (edge, head); lane covers 8 of the 256 channels
__global__ void k_edge_agg(const float* __restrict__ eBuf, const float* __restrict__ den,
                           const int* __restrict__ srcA, const int* __restrict__ dstA,
                           const unsigned short* __restrict__ hMat,
                           float* __restrict__ agg,
                           int E, int Ntot, int H, int C)
{
    long long gid = blockIdx.x * (long long)blockDim.x + threadIdx.x;
    long long wid = gid >> 5;
    int lane = (int)(gid & 31);
    long long tot = (long long)(E + Ntot) * H;
    if (wid >= tot) return;
    int i  = (int)(wid / H), hh = (int)(wid % H);
    int s  = (i < E) ? srcA[i] : (i - E);
    int d  = (i < E) ? dstA[i] : (i - E);
    float alpha = eBuf[wid] / (den[d * H + hh] + 1e-16f);
    const unsigned short* hp = hMat + (size_t)s * H * C + (size_t)hh * C + lane * 8;
    float* op = agg + (size_t)d * H * C + (size_t)hh * C + lane * 8;
    uint4 q = *(const uint4*)hp;
    unsigned int w[4] = {q.x, q.y, q.z, q.w};
#pragma unroll
    for (int j = 0; j < 4; ++j) {
        atomicAdd(op + 2*j,     alpha * bfbits2f(w[j] & 0xFFFFu));
        atomicAdd(op + 2*j + 1, alpha * bfbits2f(w[j] >> 16));
    }
}

// ---------------- epilogue: x = bf16(relu(agg + bias)) ---------------------
__global__ void k_bias_relu_cvt(const float* __restrict__ agg, const float* __restrict__ bias,
                                unsigned short* __restrict__ out, long long total, int HC)
{
    long long i = blockIdx.x * (long long)blockDim.x + threadIdx.x;
    if (i >= total) return;
    int col = (int)(i % HC);
    float v = agg[i] + bias[col];
    out[i] = f2bf(fmaxf(v, 0.0f));
}

// ---------------------------------------------------------------------------
extern "C" void kernel_launch(void* const* d_in, const int* in_sizes, int n_in,
                              void* d_out, int out_size, void* d_ws, size_t ws_size,
                              hipStream_t stream) {
    const float* constraints = (const float*)d_in[0];
    const float* columns     = (const float*)d_in[1];
    const int*   edges       = (const int*)d_in[2];
    const float* W_node = (const float*)d_in[3];
    const float* b_node = (const float*)d_in[4];
    const float* W_col  = (const float*)d_in[5];
    const float* b_col  = (const float*)d_in[6];
    const float* W1     = (const float*)d_in[7];
    const float* attS1  = (const float*)d_in[8];
    const float* attD1  = (const float*)d_in[9];
    const float* b1     = (const float*)d_in[10];
    const float* W2     = (const float*)d_in[11];
    const float* attS2  = (const float*)d_in[12];
    const float* attD2  = (const float*)d_in[13];
    const float* b2     = (const float*)d_in[14];
    const float* W_out  = (const float*)d_in[15];
    const float* b_out  = (const float*)d_in[16];

    const int NF = 64, CF = 128, HID = 256, H1 = 8, EMB = 128;
    const int HC1  = H1 * HID;                 // 2048
    const int Nc   = in_sizes[0] / NF;         // 4000
    const int Ncol = in_sizes[1] / CF;         // 16000
    const int E    = in_sizes[2] / 2;          // 100000
    const int Ntot = Nc + Ncol;                // 20000
    const int Etot = E + Ntot;                 // 120000
    const int* e0 = edges;                     // edges[0]
    const int* e1 = edges + E;                 // edges[1]

    char* ws = (char*)d_ws;
    size_t off = 0;
    auto alloc = [&](size_t bytes) { size_t o = off; off = (off + bytes + 255) & ~(size_t)255; return o; };

    size_t o_WnF = alloc((size_t)HID * NF  * 2);   // folded W_node, bf16
    size_t o_WcF = alloc((size_t)HID * CF  * 2);   // folded W_col, bf16
    size_t o_W1b = alloc((size_t)HC1 * HID * 2);
    size_t o_W2b = alloc((size_t)HID * HC1 * 2);
    size_t o_Wob = alloc((size_t)EMB * HID * 2);
    size_t o_cb  = alloc((size_t)Nc   * NF  * 2);  // constraints bf16
    size_t o_clb = alloc((size_t)Ncol * CF  * 2);  // columns bf16
    size_t o_x0  = alloc((size_t)Ntot * HID * 2);  // embedded x, bf16
    size_t o_h1  = alloc((size_t)Ntot * HC1 * 2);  // conv1 proj (later reused as x1)
    size_t o_aS  = alloc((size_t)Ntot * H1 * 4);
    size_t o_aD  = alloc((size_t)Ntot * H1 * 4);
    size_t o_m   = alloc((size_t)Ntot * H1 * 4);
    size_t o_den = alloc((size_t)Ntot * H1 * 4);
    size_t o_e   = alloc((size_t)Etot * H1 * 4);
    size_t o_agg = alloc((size_t)Ntot * HC1 * 4);  // f32 accumulator (reused conv2)
    size_t o_h2  = alloc((size_t)Ntot * HID * 2);
    size_t o_x2  = alloc((size_t)Ntot * HID * 2);
    (void)ws_size; (void)n_in; (void)out_size;

    unsigned short* WnF = (unsigned short*)(ws + o_WnF);
    unsigned short* WcF = (unsigned short*)(ws + o_WcF);
    unsigned short* W1b = (unsigned short*)(ws + o_W1b);
    unsigned short* W2b = (unsigned short*)(ws + o_W2b);
    unsigned short* Wob = (unsigned short*)(ws + o_Wob);
    unsigned short* cb  = (unsigned short*)(ws + o_cb);
    unsigned short* clb = (unsigned short*)(ws + o_clb);
    unsigned short* x0  = (unsigned short*)(ws + o_x0);
    unsigned short* h1  = (unsigned short*)(ws + o_h1);
    unsigned short* x1  = (unsigned short*)(ws + o_h1);   // reuse after aggregation
    float* aS  = (float*)(ws + o_aS);
    float* aD  = (float*)(ws + o_aD);
    unsigned int* mK = (unsigned int*)(ws + o_m);
    float* den = (float*)(ws + o_den);
    float* eB  = (float*)(ws + o_e);
    float* agg = (float*)(ws + o_agg);
    unsigned short* h2 = (unsigned short*)(ws + o_h2);
    unsigned short* x2 = (unsigned short*)(ws + o_x2);

    const int B = 256;
    auto nb = [&](long long n) { return (unsigned)((n + B - 1) / B); };
    auto gemm = [&](const unsigned short* A, const unsigned short* Wb, const float* bias,
                    float* outF, unsigned short* outB, int M, int N, int K, int flags) {
        dim3 g((unsigned)(N / 64), (unsigned)((M / 32 + 3) / 4));
        k_gemm_bf16<<<g, 128, 0, stream>>>(A, Wb, bias, outF, outB, M, N, K, flags);
    };

    // ---- weight / input conversion -------------------------------------
    k_fold_w<<<nb(HID * NF), B, 0, stream>>>(W_node, WnF, HID, NF);
    k_fold_w<<<nb(HID * CF), B, 0, stream>>>(W_col,  WcF, HID, CF);
    k_cvt_bf16<<<nb((long long)HC1 * HID), B, 0, stream>>>(W1,    W1b, (long long)HC1 * HID);
    k_cvt_bf16<<<nb((long long)HID * HC1), B, 0, stream>>>(W2,    W2b, (long long)HID * HC1);
    k_cvt_bf16<<<nb((long long)EMB * HID), B, 0, stream>>>(W_out, Wob, (long long)EMB * HID);
    k_cvt_bf16<<<nb((long long)Nc   * NF), B, 0, stream>>>(constraints, cb,  (long long)Nc * NF);
    k_cvt_bf16<<<nb((long long)Ncol * CF), B, 0, stream>>>(columns,     clb, (long long)Ncol * CF);

    // ---- embed: x = relu([c;col] @ Wfold^T + b) ------------------------
    gemm(cb,  WnF, b_node, nullptr, x0,                         Nc,   HID, NF, GF_BIAS | GF_RELU | GF_BF16);
    gemm(clb, WcF, b_col,  nullptr, x0 + (size_t)Nc * HID,      Ncol, HID, CF, GF_BIAS | GF_RELU | GF_BF16);

    // ---- conv1 (H=8) ---------------------------------------------------
    gemm(x0, W1b, nullptr, nullptr, h1, Ntot, HC1, HID, GF_BF16);
    k_att_dots<<<nb(Ntot * H1), B, 0, stream>>>(h1, attS1, attD1, aS, aD, Ntot, H1, HID);
    k_fill0<<<nb((long long)Ntot * H1), B, 0, stream>>>(mK, (long long)Ntot * H1);
    k_fill0<<<nb((long long)Ntot * H1), B, 0, stream>>>((unsigned int*)den, (long long)Ntot * H1);
    k_fill0<<<nb((long long)Ntot * HC1), B, 0, stream>>>((unsigned int*)agg, (long long)Ntot * HC1);
    long long tot1 = (long long)Etot * H1;
    k_edge_max   <<<nb(tot1), B, 0, stream>>>(aS, aD, e0, e1, E, Ntot, H1, eB, mK);
    k_edge_expsum<<<nb(tot1), B, 0, stream>>>(e0, e1, E, Ntot, H1, mK, eB, den);
    k_edge_agg   <<<nb(tot1 * 32), B, 0, stream>>>(eB, den, e0, e1, h1, agg, E, Ntot, H1, HID);
    k_bias_relu_cvt<<<nb((long long)Ntot * HC1), B, 0, stream>>>(agg, b1, x1, (long long)Ntot * HC1, HC1);

    // ---- conv2 (H=1, flipped edges) ------------------------------------
    gemm(x1, W2b, nullptr, nullptr, h2, Ntot, HID, HC1, GF_BF16);
    k_att_dots<<<nb(Ntot), B, 0, stream>>>(h2, attS2, attD2, aS, aD, Ntot, 1, HID);
    k_fill0<<<nb((long long)Ntot), B, 0, stream>>>(mK, (long long)Ntot);
    k_fill0<<<nb((long long)Ntot), B, 0, stream>>>((unsigned int*)den, (long long)Ntot);
    k_fill0<<<nb((long long)Ntot * HID), B, 0, stream>>>((unsigned int*)agg, (long long)Ntot * HID);
    long long tot2 = (long long)Etot;
    k_edge_max   <<<nb(tot2), B, 0, stream>>>(aS, aD, e1, e0, E, Ntot, 1, eB, mK);
    k_edge_expsum<<<nb(tot2), B, 0, stream>>>(e1, e0, E, Ntot, 1, mK, eB, den);
    k_edge_agg   <<<nb(tot2 * 32), B, 0, stream>>>(eB, den, e1, e0, h2, agg, E, Ntot, 1, HID);
    k_bias_relu_cvt<<<nb((long long)Ntot * HID), B, 0, stream>>>(agg, b2, x2, (long long)Ntot * HID, HID);

    // ---- logits: x2[Nc:] @ W_out^T + b_out -> d_out (f32) --------------
    gemm(x2 + (size_t)Nc * HID, Wob, b_out, (float*)d_out, nullptr, Ncol, EMB, HID, GF_BIAS);
}
